// Module_9740985828114
// MI455X (gfx1250) — matrix operand order repaired and checked by reading, not compiled, not run on hardware
//
#include <hip/hip_runtime.h>
#include <hip/hip_bf16.h>

typedef __attribute__((ext_vector_type(16))) __bf16 bf16x16;
typedef __attribute__((ext_vector_type(8)))  __bf16 bf16x8;
typedef __attribute__((ext_vector_type(8)))  float  f32x8;

#define B_   2
#define T_   1024
#define SC_  1024
#define D_   2048
#define N_   16
#define KH_  4
#define H_   128
#define S_   2048
#define SOFTCAP_ 50.0f
#define BIG_NEG_ -2.3819763e38f
#define LN1E4_ 9.210340371976184f   /* ln(10000) */

static __device__ __forceinline__ bf16x16 cat8(bf16x8 lo, bf16x8 hi) {
  return __builtin_shufflevector(lo, hi, 0,1,2,3,4,5,6,7,8,9,10,11,12,13,14,15);
}
static __device__ __forceinline__ f32x8 wmma_bf16(bf16x16 a, bf16x16 b, f32x8 c) {
  // D = A(16x32 bf16) * B(32x16 bf16) + C(16x16 f32)
  return __builtin_amdgcn_wmma_f32_16x16x32_bf16(false, a, false, b, (short)0, c,
                                                 false, false);
}
static __device__ __forceinline__ void wave_lds_fence() {
#if __has_builtin(__builtin_amdgcn_s_wait_dscnt)
  __builtin_amdgcn_s_wait_dscnt(0);
#else
  asm volatile("s_wait_dscnt 0" ::: "memory");
#endif
}
static __device__ __forceinline__ void wave_async_fence() {
#if __has_builtin(__builtin_amdgcn_s_wait_asynccnt)
  __builtin_amdgcn_s_wait_asynccnt(0);
#else
  asm volatile("s_wait_asynccnt 0" ::: "memory");
#endif
}
/* Async DMA: 16 bytes per lane, global -> LDS, tracked by ASYNCcnt.
 * LDS operand is the low 32 bits of the generic shared pointer (aperture
 * mapping truncates high bits to reach the wave's LDS window). */
static __device__ __forceinline__ void async_copy_b128(void* lds_ptr,
                                                       const void* gptr) {
  uint32_t lds_off = (uint32_t)(uintptr_t)lds_ptr;
  asm volatile("global_load_async_to_lds_b128 %0, %1, off"
               :: "v"(lds_off), "v"(gptr) : "memory");
}

/* ------------------------------------------------------------------ */
/* Kernel 1: fp32 KV cache [B,SC,K,H] -> bf16 [B,K,S,H] (head-major)  */
/* ------------------------------------------------------------------ */
__global__ __launch_bounds__(256) void k_cache_cast(
    const float* __restrict__ ck, const float* __restrict__ cv,
    __bf16* __restrict__ kb, __bf16* __restrict__ vb) {
  int idx = blockIdx.x * 256 + threadIdx.x;          // B*SC*K*H = 1M elems
  int h  = idx & (H_ - 1);
  int kh = (idx >> 7) & (KH_ - 1);
  int s  = (idx >> 9) & (SC_ - 1);
  int b  = idx >> 19;
  size_t dst = (((size_t)b * KH_ + kh) * S_ + s) * H_ + h;
  kb[dst] = (__bf16)ck[idx];
  vb[dst] = (__bf16)cv[idx];
}

/* ------------------------------------------------------------------ */
/* Kernel 2: fused QKV projection + RoPE.                              */
/* Logical GEMM: X(2048 x 2048) * W(2048 x 24*128).                   */
/* grid = (16 row tiles of 128, 24 slots). Slot = one 128-wide head.  */
/* Wave w owns rows 16w..16w+15 and all 128 cols -> 8 accumulators.   */
/* ------------------------------------------------------------------ */
__global__ __launch_bounds__(256) void k_qkv(
    const float* __restrict__ x, const float* __restrict__ wq,
    const float* __restrict__ wkv,
    __bf16* __restrict__ qb, __bf16* __restrict__ kb, __bf16* __restrict__ vb) {
  __shared__ __align__(16) __bf16 Xs[128 * 32];   // [row][k]
  __shared__ __align__(16) __bf16 Wt[128 * 32];   // transposed: [col][k]
  const int tid  = threadIdx.x;
  const int lane = tid & 31, w = tid >> 5;
  const int half = lane >> 4, nidx = lane & 15;
  const int row0 = blockIdx.x * 128;
  const int slot = blockIdx.y;                    // 0..15 q | 16..19 k | 20..23 v
  const float* wbase = (slot < 16) ? (wq + (size_t)slot * (D_ * H_))
                                   : (wkv + (size_t)(slot - 16) * (D_ * H_));
  f32x8 acc[8];
#pragma unroll
  for (int j = 0; j < 8; ++j) acc[j] = (f32x8){0,0,0,0,0,0,0,0};

  for (int k0 = 0; k0 < D_; k0 += 32) {
    if (k0 + 32 < D_)  // CDNA5 prefetch of next X tile -> global_prefetch_b8
      __builtin_prefetch(x + (size_t)(row0 + (tid >> 1)) * D_ + k0 + 32);
#pragma unroll
    for (int e = 0; e < 4; ++e) {                 // stage X (128x32 f32->bf16)
      int g = tid + 256 * e;
      int rr = g >> 3, c4 = (g & 7) << 2;
      const float4 v = *(const float4*)(x + (size_t)(row0 + rr) * D_ + k0 + c4);
      Xs[rr * 32 + c4 + 0] = (__bf16)v.x;
      Xs[rr * 32 + c4 + 1] = (__bf16)v.y;
      Xs[rr * 32 + c4 + 2] = (__bf16)v.z;
      Xs[rr * 32 + c4 + 3] = (__bf16)v.w;
    }
#pragma unroll
    for (int e = 0; e < 4; ++e) {                 // stage W transposed (32x128)
      int g = tid + 256 * e;
      int kr = g >> 5, c4 = (g & 31) << 2;
      const float4 v = *(const float4*)(wbase + (size_t)(k0 + kr) * H_ + c4);
      Wt[(c4 + 0) * 32 + kr] = (__bf16)v.x;
      Wt[(c4 + 1) * 32 + kr] = (__bf16)v.y;
      Wt[(c4 + 2) * 32 + kr] = (__bf16)v.z;
      Wt[(c4 + 3) * 32 + kr] = (__bf16)v.w;
    }
    __syncthreads();
    const __bf16* arow = Xs + (w * 16 + nidx) * 32;   // A frag: lane = row m
    bf16x16 a = cat8(*(const bf16x8*)(arow + half * 8),
                     *(const bf16x8*)(arow + 16 + half * 8));
#pragma unroll
    for (int j = 0; j < 8; ++j) {
      const __bf16* bp = Wt + (16 * j + nidx) * 32 + half * 8;  // table order: 8h.., 16+8h..
      bf16x16 bfr = cat8(*(const bf16x8*)bp, *(const bf16x8*)(bp + 16));
      acc[j] = wmma_bf16(a, bfr, acc[j]);
    }
    __syncthreads();
  }

  /* epilogue: C frag (m,n): lane<16 -> m=r,n=lane ; lane>=16 -> m=8+r,n=lane-16 */
  const int rbase = row0 + 16 * w;
  const int b    = rbase >> 10;        // T_ = 1024, tile never crosses batch
  const int tloc = rbase & (T_ - 1);
  if (slot < 20) {                     // q or k: apply RoPE
    const bool isq  = (slot < 16);
    const float osc = isq ? 0.08838834764831845f : 1.0f;   // h^-0.5 for q
    __bf16* dst = isq
        ? (qb + (((size_t)b * N_ + slot) * T_ + tloc) * H_)
        : (kb + (((size_t)b * KH_ + (slot - 16)) * S_ + SC_ + tloc) * H_);
#pragma unroll
    for (int j = 0; j < 4; ++j) {
      int h = 16 * j + nidx;                         // h in [0,64)
      float inv_ts = __expf(-(float)h * (LN1E4_ / 64.0f));  // 10000^(-h/64)
#pragma unroll
      for (int r = 0; r < 8; ++r) {
        int m = r + 8 * half;
        float pos = (float)(SC_ + tloc + m);
        float sn, cs;
        __sincosf(pos * inv_ts, &sn, &cs);
        float x1 = acc[j][r], x2 = acc[j + 4][r];
        dst[(size_t)m * H_ + h]      = (__bf16)((x1 * cs - x2 * sn) * osc);
        dst[(size_t)m * H_ + h + 64] = (__bf16)((x2 * cs + x1 * sn) * osc);
      }
    }
  } else {                             // v: plain store
    __bf16* dst = vb + (((size_t)b * KH_ + (slot - 20)) * S_ + SC_ + tloc) * H_;
#pragma unroll
    for (int j = 0; j < 8; ++j)
#pragma unroll
      for (int r = 0; r < 8; ++r) {
        int m = r + 8 * half;
        dst[(size_t)m * H_ + 16 * j + nidx] = (__bf16)acc[j][r];
      }
  }
}

/* ------------------------------------------------------------------ */
/* Kernel 3: flash attention. grid = (8 q-tiles, B*N).                */
/* Wave w: 16 queries. S-loop in tiles of 64 keys with online softmax.*/
/* K tile staged via async global->LDS DMA (ASYNCcnt).                */
/* ------------------------------------------------------------------ */
__global__ __launch_bounds__(256) void k_attn(
    const __bf16* __restrict__ qb, const __bf16* __restrict__ kb,
    const __bf16* __restrict__ vb, __bf16* __restrict__ enc) {
  __shared__ __align__(16) __bf16 Ks[64 * 128];     // [s][h]
  __shared__ __align__(16) __bf16 Vt[128 * 64];     // transposed [h][s]
  __shared__ __align__(16) __bf16 Ps[8 * 16 * 64];  // per-wave P scratch
  const int tid  = threadIdx.x;
  const int lane = tid & 31, w = tid >> 5;
  const int half = lane >> 4, nidx = lane & 15;
  const int qtile = blockIdx.x;
  const int bh = blockIdx.y;
  const int b  = bh >> 4;
  const int nh = bh & 15;
  const int kh = nh >> 2;                       // G = 4 query heads per kv head
  const int t0 = qtile * 128 + w * 16;

  const __bf16* qbase = qb + (((size_t)b * N_ + nh) * T_ + t0) * H_;
  bf16x16 aq[4];
#pragma unroll
  for (int f = 0; f < 4; ++f) {                 // Q A-frags, H = 4 x 32
    const __bf16* qr = qbase + (size_t)nidx * H_ + 32 * f;
    aq[f] = cat8(*(const bf16x8*)(qr + half * 8),
                 *(const bf16x8*)(qr + 16 + half * 8));
  }
  float mrow[8], lrow[8];
  f32x8 o[8];
#pragma unroll
  for (int r = 0; r < 8; ++r) { mrow[r] = BIG_NEG_; lrow[r] = 0.0f; }
#pragma unroll
  for (int j = 0; j < 8; ++j) o[j] = (f32x8){0,0,0,0,0,0,0,0};

  const __bf16* kbase = kb + (((size_t)b * KH_ + kh) * S_) * H_;
  const __bf16* vbase = vb + (((size_t)b * KH_ + kh) * S_) * H_;
  const int s_end = SC_ + qtile * 128 + 128;    // block-uniform causal bound

  for (int ks = 0; ks < s_end; ks += 64) {
#pragma unroll
    for (int e = 0; e < 4; ++e) {               // stage K (async DMA) + V^T
      int g = tid + 256 * e;
      int sr = g >> 4, c8 = (g & 15) << 3;
      async_copy_b128(Ks + sr * 128 + c8,
                      kbase + (size_t)(ks + sr) * H_ + c8);
      bf16x8 vv = *(const bf16x8*)(vbase + (size_t)(ks + sr) * H_ + c8);
#pragma unroll
      for (int i = 0; i < 8; ++i) Vt[(c8 + i) * 64 + sr] = vv[i];
    }
    wave_async_fence();                         // my K copies landed in LDS
    __syncthreads();                            // everyone's copies visible
    if (ks <= SC_ + t0 + 15) {                  // wave-uniform skip
      f32x8 lg[4];
#pragma unroll
      for (int j = 0; j < 4; ++j) {             // logits = Q * K^T (16x64)
        lg[j] = (f32x8){0,0,0,0,0,0,0,0};
#pragma unroll
        for (int f = 0; f < 4; ++f) {
          const __bf16* bp = Ks + (16 * j + nidx) * 128 + 32 * f + half * 8;
          bf16x16 bk = cat8(*(const bf16x8*)bp, *(const bf16x8*)(bp + 16));
          lg[j] = wmma_bf16(aq[f], bk, lg[j]);
        }
      }
#pragma unroll
      for (int r = 0; r < 8; ++r) {             // softcap + mask + online softmax
        int t = t0 + r + 8 * half;
        float mx = BIG_NEG_;
#pragma unroll
        for (int j = 0; j < 4; ++j) {
          int s = ks + 16 * j + nidx;
          float v = tanhf(lg[j][r] * (1.0f / SOFTCAP_)) * SOFTCAP_;
          if (s > SC_ + t) v = BIG_NEG_;
          lg[j][r] = v;
          mx = fmaxf(mx, v);
        }
#pragma unroll
        for (int off = 8; off >= 1; off >>= 1)
          mx = fmaxf(mx, __shfl_xor(mx, off, 32));
        float mnew = fmaxf(mrow[r], mx);
        float resc = __expf(mrow[r] - mnew);
        float rs = 0.0f;
#pragma unroll
        for (int j = 0; j < 4; ++j) {
          float p = __expf(lg[j][r] - mnew);
          lg[j][r] = p;
          rs += p;
        }
#pragma unroll
        for (int off = 8; off >= 1; off >>= 1) rs += __shfl_xor(rs, off, 32);
        mrow[r] = mnew;
        lrow[r] = lrow[r] * resc + rs;
#pragma unroll
        for (int jh = 0; jh < 8; ++jh) o[jh][r] *= resc;
      }
      /* P: C-frag layout -> A-frag layout via per-wave LDS round trip */
      __bf16* pw = Ps + w * (16 * 64);
#pragma unroll
      for (int j = 0; j < 4; ++j)
#pragma unroll
        for (int r = 0; r < 8; ++r)
          pw[(r + 8 * half) * 64 + 16 * j + nidx] = (__bf16)lg[j][r];
      wave_lds_fence();                         // wave-local DS ordering
      bf16x16 pa[2];
#pragma unroll
      for (int pf = 0; pf < 2; ++pf) {
        const __bf16* pr = pw + nidx * 64 + 32 * pf;
        pa[pf] = cat8(*(const bf16x8*)(pr + half * 8),
                      *(const bf16x8*)(pr + 16 + half * 8));
      }
#pragma unroll
      for (int jh = 0; jh < 8; ++jh)            // O += P * V
#pragma unroll
        for (int pf = 0; pf < 2; ++pf) {
          const __bf16* vp = Vt + (16 * jh + nidx) * 64 + 32 * pf + half * 8;
          bf16x16 bv = cat8(*(const bf16x8*)vp, *(const bf16x8*)(vp + 16));
          o[jh] = wmma_bf16(pa[pf], bv, o[jh]);
        }
    }
    __syncthreads();
  }
  float inv[8];
#pragma unroll
  for (int r = 0; r < 8; ++r) inv[r] = 1.0f / lrow[r];
  __bf16* ebase = enc + ((size_t)b * T_ + t0) * (N_ * H_) + nh * H_;
#pragma unroll
  for (int jh = 0; jh < 8; ++jh)
#pragma unroll
    for (int r = 0; r < 8; ++r) {
      int m = r + 8 * half;
      ebase[(size_t)m * (N_ * H_) + 16 * jh + nidx] = (__bf16)(o[jh][r] * inv[r]);
    }
}

/* ------------------------------------------------------------------ */
/* Kernel 4: out = enc(2048x2048 bf16) * wout(2048x2048 f32->bf16)    */
/* grid = (16 row tiles, 16 col tiles)                                */
/* ------------------------------------------------------------------ */
__global__ __launch_bounds__(256) void k_out(
    const __bf16* __restrict__ enc, const float* __restrict__ wout,
    float* __restrict__ out) {
  __shared__ __align__(16) __bf16 Xs[128 * 32];
  __shared__ __align__(16) __bf16 Wt[128 * 32];   // transposed [col][k]
  const int tid  = threadIdx.x;
  const int lane = tid & 31, w = tid >> 5;
  const int half = lane >> 4, nidx = lane & 15;
  const int row0 = blockIdx.x * 128;
  const int col0 = blockIdx.y * 128;
  f32x8 acc[8];
#pragma unroll
  for (int j = 0; j < 8; ++j) acc[j] = (f32x8){0,0,0,0,0,0,0,0};

  for (int k0 = 0; k0 < N_ * H_; k0 += 32) {
    if (k0 + 32 < N_ * H_)
      __builtin_prefetch(wout + (size_t)(k0 + 32 + (tid >> 3)) * D_ + col0);
#pragma unroll
    for (int e = 0; e < 2; ++e) {               // stage enc tile (bf16 direct)
      int g = tid + 256 * e;
      int rr = g >> 2, c8 = (g & 3) << 3;
      *(bf16x8*)(Xs + rr * 32 + c8) =
          *(const bf16x8*)(enc + (size_t)(row0 + rr) * (N_ * H_) + k0 + c8);
    }
#pragma unroll
    for (int e = 0; e < 4; ++e) {               // stage wout transposed
      int g = tid + 256 * e;
      int kr = g >> 5, c4 = (g & 31) << 2;
      const float4 v = *(const float4*)(wout + (size_t)(k0 + kr) * D_ + col0 + c4);
      Wt[(c4 + 0) * 32 + kr] = (__bf16)v.x;
      Wt[(c4 + 1) * 32 + kr] = (__bf16)v.y;
      Wt[(c4 + 2) * 32 + kr] = (__bf16)v.z;
      Wt[(c4 + 3) * 32 + kr] = (__bf16)v.w;
    }
    __syncthreads();
    const __bf16* arow = Xs + (w * 16 + nidx) * 32;
    bf16x16 a = cat8(*(const bf16x8*)(arow + half * 8),
                     *(const bf16x8*)(arow + 16 + half * 8));
#pragma unroll
    for (int j = 0; j < 8; ++j) {
      const __bf16* bp = Wt + (16 * j + nidx) * 32 + half * 8;
      bf16x16 bfr = cat8(*(const bf16x8*)bp, *(const bf16x8*)(bp + 16));
      acc[j] = wmma_bf16(a, bfr, acc[j]);
    }
    __syncthreads();
  }
#pragma unroll
  for (int j = 0; j < 8; ++j)
#pragma unroll
    for (int r = 0; r < 8; ++r) {
      int m = r + 8 * half;
      out[(size_t)(row0 + w * 16 + m) * D_ + col0 + 16 * j + nidx] = acc[j][r];
    }
}

/* ------------------------------------------------------------------ */
extern "C" void kernel_launch(void* const* d_in, const int* in_sizes, int n_in,
                              void* d_out, int out_size, void* d_ws, size_t ws_size,
                              hipStream_t stream) {
  const float* x    = (const float*)d_in[0];
  /* d_in[1] positions, d_in[2] attn_mask: implied by indices, unused */
  const float* ck   = (const float*)d_in[3];
  const float* cv   = (const float*)d_in[4];
  const float* wq   = (const float*)d_in[5];
  const float* wkv  = (const float*)d_in[6];
  const float* wout = (const float*)d_in[7];
  float* out = (float*)d_out;

  char* ws = (char*)d_ws;
  __bf16* qb  = (__bf16*)(ws);                      // [B,N,T,H]   8 MB
  __bf16* kb  = (__bf16*)(ws + (size_t)(8  << 20)); // [B,K,S,H]   4 MB
  __bf16* vb  = (__bf16*)(ws + (size_t)(12 << 20)); // [B,K,S,H]   4 MB
  __bf16* enc = (__bf16*)(ws + (size_t)(16 << 20)); // [B,T,N*H]   8 MB

  k_cache_cast<<<dim3((B_ * SC_ * KH_ * H_) / 256), 256, 0, stream>>>(ck, cv, kb, vb);
  k_qkv<<<dim3(16, 24), 256, 0, stream>>>(x, wq, wkv, qb, kb, vb);
  k_attn<<<dim3(8, B_ * N_), 256, 0, stream>>>(qb, kb, vb, enc);
  k_out<<<dim3(16, 16), 256, 0, stream>>>(enc, wout, out);
}
